// self_Attention_79920751444426
// MI455X (gfx1250) — matrix operand order repaired and checked by reading, not compiled, not run on hardware
//
#include <hip/hip_runtime.h>
#include <hip/hip_bf16.h>

#define INPUT_DIM 1024
#define HID       1024
#define S_LEN     2048
#define BATCH     4
#define KT        128   // key tile in attention
#define NWAVES    8     // waves per attention block

typedef __attribute__((ext_vector_type(16))) __bf16       v16bf;
typedef __attribute__((ext_vector_type(8)))  float        v8f;
typedef __attribute__((ext_vector_type(4)))  unsigned int uint4v;

union ABu   { uint4v u[2]; v16bf v; };          // 32B fragment (8 VGPRs)
union Pack8 { unsigned short s[8]; uint4v u; }; // 8 bf16 packed for 16B store

__device__ __forceinline__ unsigned short f32_to_bf16(float f) {
  unsigned int u = __float_as_uint(f);
  u += 0x7FFFu + ((u >> 16) & 1u);   // round-to-nearest-even
  return (unsigned short)(u >> 16);
}

// ---------------------------------------------------------------------------
// fp32 -> bf16 elementwise conversion (grid-stride)
// ---------------------------------------------------------------------------
__global__ void cvt_bf16_kernel(const float* __restrict__ src,
                                unsigned short* __restrict__ dst, int n) {
  int i = blockIdx.x * blockDim.x + threadIdx.x;
  int stride = gridDim.x * blockDim.x;
  for (; i < n; i += stride) dst[i] = f32_to_bf16(src[i]);
}

// ---------------------------------------------------------------------------
// QKV projection: C[m,n] = sum_d X[m,d] * W[n,d] + bias[n]
// X bf16 [B*S][D] row-major, W bf16 [H][D] row-major.
// transposeV==0: store bf16 row-major [B*S][H]
// transposeV==1: store bf16 transposed [B][H][S]   (for attn P@V B-frags)
// Block: 128 threads = 4 waves, tile 64(M) x 64(N); wave w owns 16 N-columns.
// ---------------------------------------------------------------------------
__global__ __launch_bounds__(128)
void qkv_gemm_kernel(const unsigned short* __restrict__ Xb,
                     const unsigned short* __restrict__ Wb,
                     const float* __restrict__ bias,
                     unsigned short* __restrict__ dst,
                     int transposeV) {
  const int lane = threadIdx.x & 31;
  const int wave = threadIdx.x >> 5;
  const int col  = lane & 15;
  const int hh   = lane >> 4;          // half-wave selector
  const int mbase = blockIdx.x * 64;
  const int n     = blockIdx.y * 64 + wave * 16 + col;  // hidden column

  v8f acc[4];
  for (int t = 0; t < 4; ++t)
    for (int v = 0; v < 8; ++v) acc[t][v] = 0.0f;

  const unsigned short* wrow = Wb + (size_t)n * INPUT_DIM;

  for (int kk = 0; kk < INPUT_DIM; kk += 32) {
    // B-frag: element i <-> K = 16*hh + i, fixed column row of W (contiguous)
    ABu bfr;
    bfr.u[0] = *(const uint4v*)(wrow + kk + 16 * hh);
    bfr.u[1] = *(const uint4v*)(wrow + kk + 16 * hh + 8);
    for (int t = 0; t < 4; ++t) {
      // A-frag: row M = lane&15; K = 16h + i (contiguous)
      const unsigned short* xrow =
          Xb + ((size_t)(mbase + t * 16 + col)) * INPUT_DIM + kk;
      ABu afr;
      afr.u[0] = *(const uint4v*)(xrow + 16 * hh);
      afr.u[1] = *(const uint4v*)(xrow + 16 * hh + 8);
      acc[t] = __builtin_amdgcn_wmma_f32_16x16x32_bf16(
          false, afr.v, false, bfr.v, (short)0, acc[t], false, false);
    }
  }

  const float bval = bias[n];
  if (!transposeV) {
    for (int t = 0; t < 4; ++t)
      for (int v = 0; v < 8; ++v) {
        int m = mbase + t * 16 + v + 8 * hh;   // C row = v + 8*half
        dst[(size_t)m * HID + n] = f32_to_bf16(acc[t][v] + bval);
      }
  } else {
    for (int t = 0; t < 4; ++t) {
      int m0 = mbase + t * 16 + 8 * hh;        // rows m0..m0+7 contiguous
      int b  = m0 / S_LEN, s0 = m0 % S_LEN;
      Pack8 p;
      for (int v = 0; v < 8; ++v) p.s[v] = f32_to_bf16(acc[t][v] + bval);
      *(uint4v*)(dst + ((size_t)b * HID + n) * S_LEN + s0) = p.u;
    }
  }
}

// ---------------------------------------------------------------------------
// Flash attention: per block 16 queries of one batch; 8 waves; key tile 128.
// The 32KB Q tile is staged into LDS ONCE via the CDNA5 async global->LDS
// path (ASYNCcnt), then every score A-fragment is served from LDS for all
// 16 key-tile iterations. Wave w computes score sub-tile S[0:16][16w:16w+16],
// shares softmax stats + exp'd P (bf16) via LDS, then accumulates P @ V into
// its own 128-wide hidden slice (8 f32 16x16 accumulators).
// ---------------------------------------------------------------------------
__global__ __launch_bounds__(256)
void attn_kernel(const unsigned short* __restrict__ Qb,   // [B*S][H] bf16
                 const unsigned short* __restrict__ Kb,   // [B*S][H] bf16
                 const unsigned short* __restrict__ Vt,   // [B][H][S] bf16
                 float* __restrict__ out) {               // [B][S][H] f32
  __shared__ unsigned short Qs[16 * HID];   // 32KB staged Q tile
  __shared__ unsigned short Pl[16 * KT];    // 16 x 128 bf16 probabilities
  __shared__ float redmax[16][NWAVES];
  __shared__ float redsum[16][NWAVES];

  const int lane = threadIdx.x & 31;
  const int wave = threadIdx.x >> 5;
  const int col  = lane & 15;
  const int hh   = lane >> 4;

  const int qblk = blockIdx.x;
  const int b    = qblk / (S_LEN / 16);
  const int q0   = (qblk % (S_LEN / 16)) * 16;
  const size_t qrow0 = (size_t)b * S_LEN + q0;
  const int hbase = wave * (HID / NWAVES);  // 128-wide hidden slice

  const float SCALE = 0.03125f;             // 1/sqrt(1024)

  // ---- stage Q tile (16 x HID bf16 = 32KB) into LDS via async copies ----
  {
    const unsigned short* gq = Qb + qrow0 * HID;
    for (int off = threadIdx.x * 8; off < 16 * HID; off += 256 * 8) {
      // flat->LDS addresses truncate to addr[31:0]; low bits ARE the LDS offset
      unsigned int lds_addr = (unsigned int)(size_t)(const void*)(Qs + off);
      const unsigned short* gaddr = gq + off;
      asm volatile("global_load_async_to_lds_b128 %0, %1, off"
                   :: "v"(lds_addr), "v"(gaddr) : "memory");
    }
    asm volatile("s_wait_asynccnt 0" ::: "memory");
  }
  __syncthreads();

  v8f O[8];
  for (int t = 0; t < 8; ++t)
    for (int v = 0; v < 8; ++v) O[t][v] = 0.0f;
  float m_i[8], l_i[8];
  for (int v = 0; v < 8; ++v) { m_i[v] = -__builtin_inff(); l_i[v] = 0.0f; }

  const unsigned short* qsrow = Qs + col * HID;   // A-frag row = lane&15 (LDS)

  for (int kt = 0; kt < S_LEN; kt += KT) {
    // ---- 1) scores: wave's 16x16 tile of Q @ K^T over H=1024 ----
    v8f sacc;
    for (int v = 0; v < 8; ++v) sacc[v] = 0.0f;
    const unsigned short* kptr =
        Kb + ((size_t)b * S_LEN + kt + wave * 16 + col) * HID;
    if (kt + KT < S_LEN)  // prefetch next K tile row for this lane
      __builtin_prefetch(kptr + (size_t)KT * HID, 0, 1);
    for (int kk = 0; kk < HID; kk += 32) {
      ABu afr, bfr;
      afr.u[0] = *(const uint4v*)(qsrow + kk + 16 * hh);        // LDS
      afr.u[1] = *(const uint4v*)(qsrow + kk + 16 * hh + 8);    // LDS
      bfr.u[0] = *(const uint4v*)(kptr + kk + 16 * hh);
      bfr.u[1] = *(const uint4v*)(kptr + kk + 16 * hh + 8);
      sacc = __builtin_amdgcn_wmma_f32_16x16x32_bf16(
          false, afr.v, false, bfr.v, (short)0, sacc, false, false);
    }

    // ---- 2) local row-max over this wave's 16 columns ----
    float lm[8];
    for (int v = 0; v < 8; ++v) { sacc[v] *= SCALE; lm[v] = sacc[v]; }
    for (int off = 1; off < 16; off <<= 1)
      for (int v = 0; v < 8; ++v)
        lm[v] = fmaxf(lm[v], __shfl_xor(lm[v], off, 16));
    if (col == 0)
      for (int v = 0; v < 8; ++v) redmax[v + 8 * hh][wave] = lm[v];
    __syncthreads();

    // ---- 3) combine max, exponentiate, local sums, stage P in LDS ----
    float alpha[8];
    for (int v = 0; v < 8; ++v) {
      float g = redmax[v + 8 * hh][0];
      for (int w = 1; w < NWAVES; ++w) g = fmaxf(g, redmax[v + 8 * hh][w]);
      float mnew = fmaxf(m_i[v], g);
      alpha[v] = __expf(m_i[v] - mnew);
      m_i[v]   = mnew;
    }
    float ls[8];
    for (int v = 0; v < 8; ++v) {
      float p = __expf(sacc[v] - m_i[v]);
      sacc[v] = p; ls[v] = p;
    }
    for (int off = 1; off < 16; off <<= 1)
      for (int v = 0; v < 8; ++v) ls[v] += __shfl_xor(ls[v], off, 16);
    if (col == 0)
      for (int v = 0; v < 8; ++v) redsum[v + 8 * hh][wave] = ls[v];
    for (int v = 0; v < 8; ++v)
      Pl[(v + 8 * hh) * KT + wave * 16 + col] = f32_to_bf16(sacc[v]);
    __syncthreads();

    // ---- 4) update l, rescale O, accumulate P @ V_tile ----
    for (int v = 0; v < 8; ++v) {
      float s = 0.0f;
      for (int w = 0; w < NWAVES; ++w) s += redsum[v + 8 * hh][w];
      l_i[v] = l_i[v] * alpha[v] + s;
    }
    for (int t = 0; t < 8; ++t)
      for (int v = 0; v < 8; ++v) O[t][v] *= alpha[v];

    const unsigned short* pl = Pl + col * KT;       // A-frag row = lane&15
    for (int kk = 0; kk < KT; kk += 32) {
      ABu afr;
      afr.u[0] = *(const uint4v*)(pl + kk + 16 * hh);
      afr.u[1] = *(const uint4v*)(pl + kk + 16 * hh + 8);
      for (int t = 0; t < 8; ++t) {
        const unsigned short* vptr =
            Vt + ((size_t)b * HID + hbase + t * 16 + col) * S_LEN
               + kt + kk + 16 * hh;
        ABu bfr;
        bfr.u[0] = *(const uint4v*)(vptr);
        bfr.u[1] = *(const uint4v*)(vptr + 8);
        O[t] = __builtin_amdgcn_wmma_f32_16x16x32_bf16(
            false, afr.v, false, bfr.v, (short)0, O[t], false, false);
      }
    }
  }

  // ---- epilogue: normalize and store fp32 ----
  float inv[8];
  for (int v = 0; v < 8; ++v) inv[v] = 1.0f / l_i[v];
  for (int t = 0; t < 8; ++t)
    for (int v = 0; v < 8; ++v) {
      int r = v + 8 * hh;
      out[((size_t)b * S_LEN + q0 + r) * HID + hbase + t * 16 + col] =
          O[t][v] * inv[v];
    }
}

// ---------------------------------------------------------------------------
// Host launch
// ---------------------------------------------------------------------------
extern "C" void kernel_launch(void* const* d_in, const int* in_sizes, int n_in,
                              void* d_out, int out_size, void* d_ws, size_t ws_size,
                              hipStream_t stream) {
  const float* X  = (const float*)d_in[0];
  const float* Wq = (const float*)d_in[1];
  const float* bq = (const float*)d_in[2];
  const float* Wk = (const float*)d_in[3];
  const float* bk = (const float*)d_in[4];
  const float* Wv = (const float*)d_in[5];
  const float* bv = (const float*)d_in[6];
  float* out = (float*)d_out;
  char* ws = (char*)d_ws;

  // workspace layout (bytes)
  unsigned short* XB  = (unsigned short*)(ws + 0);                     // 16 MB
  unsigned short* WQB = (unsigned short*)(ws + (size_t)16 * (1u<<20)); // 2 MB
  unsigned short* WKB = (unsigned short*)(ws + (size_t)18 * (1u<<20)); // 2 MB
  unsigned short* WVB = (unsigned short*)(ws + (size_t)20 * (1u<<20)); // 2 MB
  unsigned short* QB  = (unsigned short*)(ws + (size_t)22 * (1u<<20)); // 16 MB
  unsigned short* KB  = (unsigned short*)(ws + (size_t)38 * (1u<<20)); // 16 MB
  unsigned short* VTB = (unsigned short*)(ws + (size_t)54 * (1u<<20)); // 16 MB

  const int NX = BATCH * S_LEN * INPUT_DIM;  // 8388608
  const int NWt = HID * INPUT_DIM;           // 1048576
  cvt_bf16_kernel<<<2048, 256, 0, stream>>>(X,  XB,  NX);
  cvt_bf16_kernel<<<512,  256, 0, stream>>>(Wq, WQB, NWt);
  cvt_bf16_kernel<<<512,  256, 0, stream>>>(Wk, WKB, NWt);
  cvt_bf16_kernel<<<512,  256, 0, stream>>>(Wv, WVB, NWt);

  dim3 g(BATCH * S_LEN / 64, HID / 64);      // 128 x 16
  qkv_gemm_kernel<<<g, 128, 0, stream>>>(XB, WQB, bq, QB,  0);
  qkv_gemm_kernel<<<g, 128, 0, stream>>>(XB, WKB, bk, KB,  0);
  qkv_gemm_kernel<<<g, 128, 0, stream>>>(XB, WVB, bv, VTB, 1);

  attn_kernel<<<BATCH * S_LEN / 16, 256, 0, stream>>>(QB, KB, VTB, out);
}
